// Channel_Self_Attn_17076789968899
// MI455X (gfx1250) — matrix operand order repaired and checked by reading, not compiled, not run on hardware
//
#include <hip/hip_runtime.h>
#include <hip/hip_bf16.h>
#include <stdint.h>

typedef __attribute__((ext_vector_type(16))) _Float16 v16h;
typedef __attribute__((ext_vector_type(8)))  float    v8f;
typedef __attribute__((ext_vector_type(4)))  _Float16 v4h;
typedef __attribute__((ext_vector_type(4)))  unsigned int v4u;

#define BB 64
#define CC 512
#define NN 1024
#define DD 64

// Async global->LDS staging (CDNA5 GLOBAL_LOAD_ASYNC_TO_LDS_B128, ASYNCcnt).
#define USE_ASYNC_LDS 1

__device__ __forceinline__ void stage_b128(_Float16* lds_dst,
                                           const _Float16* gsrc) {
#if USE_ASYNC_LDS
  uint32_t l = (uint32_t)(uintptr_t)lds_dst;
  asm volatile("global_load_async_to_lds_b128 %0, %1, off"
               :: "v"(l), "v"(gsrc) : "memory");
#else
  *(v4u*)lds_dst = *(const v4u*)gsrc;
#endif
}

// Wait until at most N async groups outstanding (in-order per wave:
// threshold N == "previous panel complete, next panel still in flight").
template <int N>
__device__ __forceinline__ void stage_wait() {
#if USE_ASYNC_LDS
  asm volatile("s_wait_asynccnt %0" :: "i"(N) : "memory");
#endif
}

// ---------------- WMMA fragment helpers (layouts per CDNA5 ISA 7.12.2) -----
// All fragment loads are exactly 2x ds_load_b128 per lane.

__device__ __forceinline__ v8f wmma_f16(v16h a, v16h b, v8f c) {
  return __builtin_amdgcn_wmma_f32_16x16x32_f16(
      /*neg_a=*/false, a, /*neg_b=*/false, b,
      /*c_mod=*/(short)0, c, /*reuse_a=*/false, /*reuse_b=*/false);
}

union Frag { v16h h; v4u u4[2]; };
union Pack8 { v4u u; _Float16 e[8]; };

// A matrix 16x32 f16 from row-major LDS: A[row][k] = base[row*ld + k]
__device__ __forceinline__ v16h load_a_frag(const _Float16* base, int ld) {
  const int lane = threadIdx.x & 31;
  const int row = lane & 15;
  const int khalf = (lane >> 4) * 8;
  const _Float16* p = base + row * ld + khalf;
  Frag f;
  f.u4[0] = *(const v4u*)(p);
  f.u4[1] = *(const v4u*)(p + 16);
  return f.h;
}

// B matrix 32x16 f16 from TRANSPOSED ([n][k]) LDS: B[k][col] = base[col*ld + k]
__device__ __forceinline__ v16h load_b_frag_t(const _Float16* base, int ld) {
  const int lane = threadIdx.x & 31;
  const int col = lane & 15;
  const int khalf = (lane >> 4) * 8;
  const _Float16* p = base + col * ld + khalf;
  Frag f;
  f.u4[0] = *(const v4u*)(p);
  f.u4[1] = *(const v4u*)(p + 16);
  return f.h;
}

__device__ __forceinline__ v8f zero_v8f() {
  v8f z = {0.f, 0.f, 0.f, 0.f, 0.f, 0.f, 0.f, 0.f};
  return z;
}

// ---------------- conversion / init --------------------------------------

__global__ void cvt_f32_f16(const float* __restrict__ in,
                            _Float16* __restrict__ out, int n4) {
  int i = blockIdx.x * blockDim.x + threadIdx.x;
  int stride = gridDim.x * blockDim.x;
  for (; i < n4; i += stride) {
    float4 f = ((const float4*)in)[i];
    v4h h = {(_Float16)f.x, (_Float16)f.y, (_Float16)f.z, (_Float16)f.w};
    ((v4h*)out)[i] = h;
  }
}

// out[c][r] = (f16) in[r][c]   (in: [R=1024][Cm] f32)
__global__ void cvt_transpose_f16(const float* __restrict__ in,
                                  _Float16* __restrict__ out, int R, int Cm) {
  int i = blockIdx.x * blockDim.x + threadIdx.x;
  if (i < R * Cm) {
    int r = i & (R - 1);
    int c = i >> 10;
    out[i] = (_Float16)in[(size_t)r * Cm + c];
  }
}

__global__ void zero_f32(float* __restrict__ p, int n) {
  int i = blockIdx.x * blockDim.x + threadIdx.x;
  if (i < n) p[i] = 0.f;
}

// ---------------- stage 1: Q = x@Qm, K = x@Km ----------------------------
// grid (4, B), 256 threads (8 waves). Block computes 128x64 of Q and K.
// Double-buffered async panels: each thread issues 4 asyncs per k-step.
__global__ __launch_bounds__(256) void qk_kernel(
    const _Float16* __restrict__ xh, const _Float16* __restrict__ qmT,
    const _Float16* __restrict__ kmT, _Float16* __restrict__ Qh,
    _Float16* __restrict__ Kh) {
  __shared__ __align__(16) _Float16 sA[2][128 * 32];
  __shared__ __align__(16) _Float16 sQT[2][64 * 32];
  __shared__ __align__(16) _Float16 sKT[2][64 * 32];
  const int b = blockIdx.y;
  const int row0 = blockIdx.x * 128;
  const int tid = threadIdx.x;
  const int wid = tid >> 5, lane = tid & 31;
  const _Float16* xb = xh + (size_t)b * CC * NN + (size_t)row0 * NN;

  auto issue = [&](int buf, int kk) {
#pragma unroll
    for (int t = tid; t < 512; t += 256) {          // x panel 128x32
      int r = t >> 2, c8 = (t & 3) * 8;
      stage_b128(sA[buf] + r * 32 + c8, xb + (size_t)r * NN + kk + c8);
    }
    int d = tid >> 2, c8 = (tid & 3) * 8;           // qmT/kmT panels 64x32
    stage_b128(sQT[buf] + d * 32 + c8, qmT + (size_t)d * NN + kk + c8);
    stage_b128(sKT[buf] + d * 32 + c8, kmT + (size_t)d * NN + kk + c8);
  };

  v8f accQ[4], accK[4];
#pragma unroll
  for (int t = 0; t < 4; ++t) { accQ[t] = zero_v8f(); accK[t] = zero_v8f(); }

  issue(0, 0);
  for (int kk = 0; kk < NN; kk += 32) {
    const int cur = (kk >> 5) & 1;
    const bool last = (kk + 32 >= NN);
    if (!last) issue(cur ^ 1, kk + 32);
    if (last) stage_wait<0>(); else stage_wait<4>();
    __syncthreads();
    v16h a = load_a_frag(sA[cur] + wid * 16 * 32, 32);
    v16h bq[4], bk[4];
#pragma unroll
    for (int t = 0; t < 4; ++t) {
      bq[t] = load_b_frag_t(sQT[cur] + t * 16 * 32, 32);
      bk[t] = load_b_frag_t(sKT[cur] + t * 16 * 32, 32);
    }
#pragma unroll
    for (int t = 0; t < 4; ++t) {
      accQ[t] = wmma_f16(a, bq[t], accQ[t]);
      accK[t] = wmma_f16(a, bk[t], accK[t]);
    }
    __syncthreads();
  }
  const int col = lane & 15, rowb = (lane >> 4) * 8;
  const size_t obase = (size_t)b * CC * DD;
#pragma unroll
  for (int t = 0; t < 4; ++t)
#pragma unroll
    for (int v = 0; v < 8; ++v) {
      int r = row0 + wid * 16 + rowb + v;
      Qh[obase + (size_t)r * DD + t * 16 + col] = (_Float16)accQ[t][v];
      Kh[obase + (size_t)r * DD + t * 16 + col] = (_Float16)accK[t][v];
    }
}

// ---------------- stage 2: Et = exp(sigmoid(Q K^T))^T, colsum -------------
// grid (4, 4, B), 256 threads. Block computes a 128x128 tile.
__global__ __launch_bounds__(256) void energy_kernel(
    const _Float16* __restrict__ Qh, const _Float16* __restrict__ Kh,
    _Float16* __restrict__ Et, float* __restrict__ colsum) {
  __shared__ __align__(16) _Float16 sQ[128 * 64];
  __shared__ __align__(16) _Float16 sK[128 * 64];
  const int b = blockIdx.z;
  const int i0 = blockIdx.x * 128, j0 = blockIdx.y * 128;
  const int tid = threadIdx.x, wid = tid >> 5, lane = tid & 31;
  const _Float16* Qb = Qh + (size_t)b * CC * DD;
  const _Float16* Kb = Kh + (size_t)b * CC * DD;

#pragma unroll
  for (int t = tid; t < 1024; t += 256) {
    int r = t >> 3, c8 = (t & 7) * 8;
    stage_b128(sQ + r * 64 + c8, Qb + (size_t)(i0 + r) * DD + c8);
    stage_b128(sK + r * 64 + c8, Kb + (size_t)(j0 + r) * DD + c8);
  }
  stage_wait<0>();
  __syncthreads();

  v8f acc[8];
#pragma unroll
  for (int t = 0; t < 8; ++t) acc[t] = zero_v8f();

#pragma unroll
  for (int ks = 0; ks < 64; ks += 32) {
    v16h a = load_a_frag(sQ + wid * 16 * 64 + ks, 64);
    v16h bt[8];
#pragma unroll
    for (int t = 0; t < 8; ++t)
      bt[t] = load_b_frag_t(sK + t * 16 * 64 + ks, 64);
#pragma unroll
    for (int t = 0; t < 8; ++t) acc[t] = wmma_f16(a, bt[t], acc[t]);
  }

  const int col = lane & 15, rowb = (lane >> 4) * 8;
  _Float16* Eb = Et + (size_t)b * CC * CC;
  const int i0l = i0 + wid * 16 + rowb;
#pragma unroll
  for (int t = 0; t < 8; ++t) {
    const int j = j0 + t * 16 + col;
    float part = 0.f;
    Pack8 pk;
#pragma unroll
    for (int v = 0; v < 8; ++v) {
      float s = acc[t][v];
      float sig = 1.0f / (1.0f + __expf(-s));
      float e = __expf(sig);  // in (1, e): softmax needs no max-shift
      part += e;
      pk.e[v] = (_Float16)e;
    }
    *(v4u*)(Eb + (size_t)j * CC + i0l) = pk.u;  // Et[j][i0l..i0l+7]
    atomicAdd(&colsum[b * CC + j], part);
  }
}

// ---------------- stage 3: fused v = W x + b, out = (E^T v) / colsum ------
// grid (16, B), 512 threads (16 waves). Block handles one 64-col strip of N.
__global__ __launch_bounds__(512) void vout_kernel(
    const _Float16* __restrict__ xh, const _Float16* __restrict__ wv,
    const float* __restrict__ bias, const _Float16* __restrict__ Et,
    const float* __restrict__ colsum, float* __restrict__ out) {
  __shared__ __align__(16) _Float16 sVT[64 * 512];     // 64 KB: v^T (f16)
  __shared__ __align__(16) _Float16 sP[2][512 * 32];   // 2x32 KB: W / Et panel
  __shared__ __align__(16) _Float16 sXT[2][64 * 32];   // 2x4 KB: x^T panel
  __shared__ float sBias[512];
  __shared__ float sRcp[512];

  const int b = blockIdx.y;
  const int n0 = blockIdx.x * 64;
  const int tid = threadIdx.x, wid = tid >> 5, lane = tid & 31;
  const int col = lane & 15, rowb = (lane >> 4) * 8;
  const int xr = tid >> 3, xc8 = (tid & 7) * 8;

  sBias[tid] = bias[tid];
  sRcp[tid] = 1.0f / colsum[b * CC + tid];

  const _Float16* xb = xh + (size_t)b * CC * NN;

  v8f acc[8];
#pragma unroll
  for (int q = 0; q < 8; ++q) acc[q] = zero_v8f();

  // ---- GEMM1: v[i][n] = sum_c W[i][c] x[c][n] ----
  Pack8 xreg;
  if (tid < 256) xreg.u = *(const v4u*)(xb + (size_t)xr * NN + n0 + xc8);
#pragma unroll
  for (int t = tid; t < 2048; t += 512) {   // prologue: W panel kk=0
    int r = t >> 2, c8 = (t & 3) * 8;
    stage_b128(sP[0] + r * 32 + c8, wv + (size_t)r * CC + c8);
  }
  for (int kk = 0; kk < CC; kk += 32) {
    const int cur = (kk >> 5) & 1;
    const bool last = (kk + 32 >= CC);
    Pack8 xnext;
    if (!last) {
#pragma unroll
      for (int t = tid; t < 2048; t += 512) {      // next W panel (async)
        int r = t >> 2, c8 = (t & 3) * 8;
        stage_b128(sP[cur ^ 1] + r * 32 + c8,
                   wv + (size_t)r * CC + (kk + 32) + c8);
      }
      if (tid < 256)                               // prefetch next x strip
        xnext.u = *(const v4u*)(xb + (size_t)(kk + 32 + xr) * NN + n0 + xc8);
    }
    if (tid < 256) {                               // scatter cur x -> x^T LDS
#pragma unroll
      for (int q = 0; q < 8; ++q) sXT[cur][(xc8 + q) * 32 + xr] = xreg.e[q];
    }
    if (last) stage_wait<0>(); else stage_wait<4>();
    __syncthreads();
    v16h a0 = load_a_frag(sP[cur] + (wid * 32) * 32, 32);
    v16h a1 = load_a_frag(sP[cur] + (wid * 32 + 16) * 32, 32);
    v16h bb[4];
#pragma unroll
    for (int t = 0; t < 4; ++t)
      bb[t] = load_b_frag_t(sXT[cur] + t * 16 * 32, 32);
#pragma unroll
    for (int t = 0; t < 4; ++t) {
      acc[t] = wmma_f16(a0, bb[t], acc[t]);
      acc[4 + t] = wmma_f16(a1, bb[t], acc[4 + t]);
    }
    if (!last) xreg = xnext;
    __syncthreads();
  }

  // v strip (+bias) -> sVT[n][i] as f16, one packed b128 store per tile
#pragma unroll
  for (int sub = 0; sub < 2; ++sub)
#pragma unroll
    for (int t = 0; t < 4; ++t) {
      int i0 = wid * 32 + sub * 16 + rowb;
      int n = t * 16 + col;
      Pack8 pk;
#pragma unroll
      for (int v = 0; v < 8; ++v)
        pk.e[v] = (_Float16)(acc[sub * 4 + t][v] + sBias[i0 + v]);
      *(v4u*)(sVT + n * 512 + i0) = pk.u;
    }
  __syncthreads();

#pragma unroll
  for (int q = 0; q < 8; ++q) acc[q] = zero_v8f();

  // ---- GEMM2 (swapped): D[row=n 64][col=j 512], K = i ----
  const int rowtile = wid >> 2;        // n sub-tile (0..3)
  const int jgroup = (wid & 3) * 128;  // 128 j columns per wave group
  const _Float16* Eb = Et + (size_t)b * CC * CC;
#pragma unroll
  for (int t = tid; t < 2048; t += 512) {   // prologue: Et panel kk=0
    int r = t >> 2, c8 = (t & 3) * 8;
    stage_b128(sP[0] + r * 32 + c8, Eb + (size_t)r * CC + c8);
  }
  for (int kk = 0; kk < CC; kk += 32) {
    const int cur = (kk >> 5) & 1;
    const bool last = (kk + 32 >= CC);
    if (!last) {
#pragma unroll
      for (int t = tid; t < 2048; t += 512) {      // next Et panel (async)
        int r = t >> 2, c8 = (t & 3) * 8;
        stage_b128(sP[cur ^ 1] + r * 32 + c8,
                   Eb + (size_t)r * CC + (kk + 32) + c8);
      }
    }
    if (last) stage_wait<0>(); else stage_wait<4>();
    __syncthreads();
    v16h a = load_a_frag(sVT + rowtile * 16 * 512 + kk, 512);
    v16h bb[8];
#pragma unroll
    for (int t = 0; t < 8; ++t)
      bb[t] = load_b_frag_t(sP[cur] + (jgroup + t * 16) * 32, 32);
#pragma unroll
    for (int t = 0; t < 8; ++t) acc[t] = wmma_f16(a, bb[t], acc[t]);
    __syncthreads();
  }

  // out[b][j][n0 + n], n consecutive per lane -> two float4 stores per tile
  float* ob = out + (size_t)b * CC * NN + n0 + rowtile * 16 + rowb;
#pragma unroll
  for (int t = 0; t < 8; ++t) {
    int j = jgroup + t * 16 + col;
    float scale = sRcp[j];
    float4 lo = {acc[t][0] * scale, acc[t][1] * scale,
                 acc[t][2] * scale, acc[t][3] * scale};
    float4 hi = {acc[t][4] * scale, acc[t][5] * scale,
                 acc[t][6] * scale, acc[t][7] * scale};
    float* p = ob + (size_t)j * NN;
    *(float4*)(p) = lo;
    *(float4*)(p + 4) = hi;
  }
}

// ---------------- host launcher ------------------------------------------

extern "C" void kernel_launch(void* const* d_in, const int* in_sizes, int n_in,
                              void* d_out, int out_size, void* d_ws,
                              size_t ws_size, hipStream_t stream) {
  const float* x  = (const float*)d_in[0];
  const float* qm = (const float*)d_in[1];
  const float* km = (const float*)d_in[2];
  const float* wv = (const float*)d_in[3];
  const float* vb = (const float*)d_in[4];
  float* out = (float*)d_out;

  char* ws = (char*)d_ws;
  size_t off = 0;
  _Float16* xh   = (_Float16*)(ws + off); off += (size_t)BB * CC * NN * 2;  // 64 MB
  _Float16* qmTh = (_Float16*)(ws + off); off += (size_t)NN * DD * 2;
  _Float16* kmTh = (_Float16*)(ws + off); off += (size_t)NN * DD * 2;
  _Float16* wvh  = (_Float16*)(ws + off); off += (size_t)CC * CC * 2;
  _Float16* Qh   = (_Float16*)(ws + off); off += (size_t)BB * CC * DD * 2;  // 4 MB
  _Float16* Kh   = (_Float16*)(ws + off); off += (size_t)BB * CC * DD * 2;  // 4 MB
  _Float16* Eth  = (_Float16*)(ws + off); off += (size_t)BB * CC * CC * 2;  // 32 MB
  float* colsum  = (float*)(ws + off);    off += (size_t)BB * CC * 4;

  cvt_f32_f16<<<2048, 256, 0, stream>>>(x, xh, BB * CC * NN / 4);
  cvt_transpose_f16<<<(NN * DD + 255) / 256, 256, 0, stream>>>(qm, qmTh, NN, DD);
  cvt_transpose_f16<<<(NN * DD + 255) / 256, 256, 0, stream>>>(km, kmTh, NN, DD);
  cvt_f32_f16<<<64, 256, 0, stream>>>(wv, wvh, CC * CC / 4);
  zero_f32<<<(BB * CC + 255) / 256, 256, 0, stream>>>(colsum, BB * CC);

  qk_kernel<<<dim3(4, BB), 256, 0, stream>>>(xh, qmTh, kmTh, Qh, Kh);
  energy_kernel<<<dim3(4, 4, BB), 256, 0, stream>>>(Qh, Kh, Eth, colsum);
  vout_kernel<<<dim3(16, BB), 512, 0, stream>>>(xh, wvh, vb, Eth, colsum, out);
}
